// MessagePassingLayer_68805376082128
// MI455X (gfx1250) — compile-verified, matrix operand order checked by reading, not run on hardware
//
#include <hip/hip_runtime.h>

#define B_ 2
#define N_ 384
#define D_ 768
#define E_ 64
#define H_ 512
#define LN_EPS 1e-5f

typedef __attribute__((ext_vector_type(16))) __bf16 v16bf;
typedef __attribute__((ext_vector_type(8)))  float  v8f;

// ---------------------------------------------------------------------------
// Pack We (=ew1[0:64]) and ew2 into bf16, laid out for the WMMA B-fragment:
// lane L holds column N = L%16 and 16 contiguous K values starting at (L/16)*16,
// so each lane reads one contiguous 32B chunk per k-step.
// packed[((nt*KT + k16)*16 + n)*16 + k] = W[(k16*16+k)*Ncols + nt*16 + n]
// ---------------------------------------------------------------------------
__global__ void k_pack(const float* __restrict__ ew1, const float* __restrict__ ew2,
                       __bf16* __restrict__ wet, __bf16* __restrict__ ew2t) {
  int idx = blockIdx.x * 256 + threadIdx.x;
  if (idx < E_ * H_) {
    // We: K=64 (k16 in 0..3), Ncols=512 (nt in 0..31)
    int k = idx & 15, n = (idx >> 4) & 15, k16 = (idx >> 8) & 3, nt = idx >> 10;
    wet[idx] = (__bf16)ew1[(size_t)(k16 * 16 + k) * H_ + nt * 16 + n];
  }
  if (idx < H_ * E_) {
    // ew2: K=512 (k16 in 0..31), Ncols=64 (nt in 0..3)
    int k = idx & 15, n = (idx >> 4) & 15, k16 = (idx >> 8) & 31, nt = idx >> 13;
    ew2t[idx] = (__bf16)ew2[(size_t)(k16 * 16 + k) * E_ + nt * 16 + n];
  }
}

// ---------------------------------------------------------------------------
// xi = nodes @ Wi, xj = nodes @ Wj  (Wi = ew1 rows 64..831, Wj = rows 832..1599)
// ---------------------------------------------------------------------------
__global__ void k_xixj(const float* __restrict__ nodes, const float* __restrict__ ew1,
                       float* __restrict__ xi, float* __restrict__ xj) {
  int row = blockIdx.x, tid = threadIdx.x;
  __shared__ float nr[D_];
  for (int d = tid; d < D_; d += 256) nr[d] = nodes[(size_t)row * D_ + d];
  __syncthreads();
  for (int h = tid; h < H_; h += 256) {
    float si = 0.f, sj = 0.f;
    for (int d = 0; d < D_; ++d) {
      float nd = nr[d];
      si += nd * ew1[(size_t)(E_ + d) * H_ + h];
      sj += nd * ew1[(size_t)(E_ + D_ + d) * H_ + h];
    }
    xi[(size_t)row * H_ + h] = si;
    xj[(size_t)row * H_ + h] = sj;
  }
}

// ---------------------------------------------------------------------------
// agg[b,i,e] = sum_j adj[b,i,j] * edges[b,i,j,e]   (bandwidth bound, 75 MB read)
// ---------------------------------------------------------------------------
__global__ void k_agg(const float* __restrict__ edges, const float* __restrict__ adj,
                      float* __restrict__ agg) {
  int bi = blockIdx.x;                       // b*N + i
  int e = threadIdx.x & 63, g = threadIdx.x >> 6;
  __shared__ float red[4][E_];
  float acc = 0.f;
  const float* erow = edges + (size_t)bi * N_ * E_;
  const float* arow = adj + (size_t)bi * N_;
  for (int j = g; j < N_; j += 4)
    acc += arow[j] * erow[(size_t)j * E_ + e];
  red[g][e] = acc;
  __syncthreads();
  if (g == 0)
    agg[(size_t)bi * E_ + e] = red[0][e] + red[1][e] + red[2][e] + red[3][e];
}

// ---------------------------------------------------------------------------
// hrelu = relu(LN(concat(agg, nodes*N) @ nw1 + nb1))
// ---------------------------------------------------------------------------
__global__ void k_node1(const float* __restrict__ agg, const float* __restrict__ nodes,
                        const float* __restrict__ nw1, const float* __restrict__ nb1,
                        const float* __restrict__ ng, const float* __restrict__ nbt,
                        float* __restrict__ hrelu) {
  int row = blockIdx.x, tid = threadIdx.x;
  __shared__ float sm[D_ + E_];
  __shared__ float red[256], red2[256];
  for (int k = tid; k < E_; k += 256) sm[k] = agg[(size_t)row * E_ + k];
  for (int k = tid; k < D_; k += 256) sm[E_ + k] = nodes[(size_t)row * D_ + k] * (float)N_;
  __syncthreads();
  float v[2];
  for (int c = 0; c < 2; ++c) {
    int h = tid + c * 256;
    float s = nb1[h];
    for (int k = 0; k < D_ + E_; ++k) s += sm[k] * nw1[(size_t)k * H_ + h];
    v[c] = s;
  }
  red[tid] = v[0] + v[1];
  red2[tid] = v[0] * v[0] + v[1] * v[1];
  __syncthreads();
  for (int o = 128; o > 0; o >>= 1) {
    if (tid < o) { red[tid] += red[tid + o]; red2[tid] += red2[tid + o]; }
    __syncthreads();
  }
  float mu = red[0] * (1.f / H_);
  float var = red2[0] * (1.f / H_) - mu * mu;
  float rs = rsqrtf(var + LN_EPS);
  for (int c = 0; c < 2; ++c) {
    int h = tid + c * 256;
    float x = (v[c] - mu) * rs * ng[h] + nbt[h];
    hrelu[(size_t)row * H_ + h] = fmaxf(x, 0.f);
  }
}

// ---------------------------------------------------------------------------
// nodes_out = nodes + hrelu @ nw2 + nb2
// ---------------------------------------------------------------------------
__global__ void k_node2(const float* __restrict__ nodes, const float* __restrict__ hrelu,
                        const float* __restrict__ nw2, const float* __restrict__ nb2,
                        float* __restrict__ out_nodes) {
  int row = blockIdx.x, tid = threadIdx.x;
  __shared__ float sh[H_];
  for (int k = tid; k < H_; k += 256) sh[k] = hrelu[(size_t)row * H_ + k];
  __syncthreads();
  for (int d = tid; d < D_; d += 256) {
    float s = nb2[d];
    for (int k = 0; k < H_; ++k) s += sh[k] * nw2[(size_t)k * D_ + d];
    out_nodes[(size_t)row * D_ + d] = nodes[(size_t)row * D_ + d] + s;
  }
}

// ---------------------------------------------------------------------------
// Fused edge MLP per tile of 32 edges (fixed b,i; N divisible by 32):
//   edge tile (32x64 f32, one contiguous 8KB block) -> LDS via async-to-LDS
//   he(32x512) = edges@We + xi + xj + eb1   [WMMA bf16; 8 waves x 64-wide H,
//                                            2 M-tiles share every B fragment]
//   LN over H=512 (shuffle + LDS-atomic cross-wave reduce), ReLU -> LDS bf16
//   out(32x64) = relu_he @ ew2              [WMMA bf16, LDS-atomic combine]
//   edges_out = edge_tile(LDS) + out + eb2
// ---------------------------------------------------------------------------
__global__ __launch_bounds__(256) void k_edge(
    const float* __restrict__ edges, const float* __restrict__ xi,
    const float* __restrict__ xj, const __bf16* __restrict__ wet,
    const __bf16* __restrict__ ew2t, const float* __restrict__ eb1,
    const float* __restrict__ eg, const float* __restrict__ ebt,
    const float* __restrict__ eb2, float* __restrict__ eout) {
  __shared__ float lsum[32], lsq[32];
  __shared__ alignas(16) float atile[32 * E_];   // raw edge tile (f32)
  __shared__ float lout[32 * E_];                // GEMM2 combine buffer
  __shared__ alignas(32) __bf16 hn[32 * H_];     // relu(LN(he)) tile

  const int tid  = threadIdx.x;
  const int wave = tid >> 5;
  const int lane = tid & 31;
  const int half = lane >> 4;   // K-half of the fragment this lane holds
  const int r    = lane & 15;   // row (A) / column (B,C,D) index within tile

  const long tileBase = (long)blockIdx.x * 32;   // flattened (b,i,j) row base
  const int  bi    = (int)(tileBase / N_);       // = b*N + i
  const int  jbase = (int)(tileBase % N_);
  const int  bN    = (bi / N_) * N_;             // = b*N

  // ---- async copy of the contiguous 32x64 f32 edge tile into LDS ----
  {
    const char* gsrc = (const char*)(edges + (size_t)tileBase * E_);
    unsigned lds0 = (unsigned)(uintptr_t)atile;
    unsigned off = (unsigned)tid * 16u;          // 256 threads x 16B = 4KB/pass
#pragma unroll
    for (int it = 0; it < 2; ++it) {
      unsigned o = off + (unsigned)it * 4096u;
      unsigned long long ga = (unsigned long long)(uintptr_t)(gsrc + o);
      unsigned la = lds0 + o;
      asm volatile("global_load_async_to_lds_b128 %0, %1, off"
                   :: "v"(la), "v"(ga) : "memory");
    }
  }

  if (tid < 32) { lsum[tid] = 0.f; lsq[tid] = 0.f; }
  for (int k = tid; k < 32 * E_; k += 256) lout[k] = 0.f;

  asm volatile("s_wait_asynccnt 0x0" ::: "memory");
  __syncthreads();

  const v8f vzero = {0.f, 0.f, 0.f, 0.f, 0.f, 0.f, 0.f, 0.f};

  // ---- GEMM1: wave handles H chunk [64*wave, +64) for both 16-row M-tiles ----
  v8f acc[2][4];
#pragma unroll
  for (int u = 0; u < 2; ++u)
#pragma unroll
    for (int t = 0; t < 4; ++t) acc[u][t] = vzero;

#pragma unroll
  for (int s = 0; s < 2; ++s) {   // K = 64 -> two k-steps of 32
    v16bf a[2];
#pragma unroll
    for (int u = 0; u < 2; ++u) {
      const float4* ap =
          (const float4*)(atile + (u * 16 + r) * E_ + 32 * s + 16 * half);
#pragma unroll
      for (int q = 0; q < 4; ++q) {
        float4 f = ap[q];
        a[u][4 * q + 0] = (__bf16)f.x; a[u][4 * q + 1] = (__bf16)f.y;
        a[u][4 * q + 2] = (__bf16)f.z; a[u][4 * q + 3] = (__bf16)f.w;
      }
    }
#pragma unroll
    for (int t = 0; t < 4; ++t) {
      int nt = wave * 4 + t;   // H n-tile 0..31
      const v16bf b = *(const v16bf*)(wet + (((nt * 4) + (2 * s + half)) * 16 + r) * 16);
#pragma unroll
      for (int u = 0; u < 2; ++u)   // B fragment reused across both M-tiles
        acc[u][t] = __builtin_amdgcn_wmma_f32_16x16x32_bf16(
            false, a[u], false, b, (short)0, acc[u][t], false, false);
    }
  }

  // ---- add xi (same for all 32 rows) + xj (per row) + eb1 ----
  const float* xrow = xi + (size_t)bi * H_;
#pragma unroll
  for (int t = 0; t < 4; ++t) {
    int h = wave * 64 + t * 16 + r;
    float base = xrow[h] + eb1[h];
#pragma unroll
    for (int u = 0; u < 2; ++u)
#pragma unroll
      for (int v = 0; v < 8; ++v) {
        int m = u * 16 + v + 8 * half;   // row within the 32-row tile
        acc[u][t][v] += base + xj[(size_t)(bN + jbase + m) * H_ + h];
      }
  }

  // ---- LayerNorm over H=512: per-row sums via shuffle + cross-wave LDS atomics
#pragma unroll
  for (int u = 0; u < 2; ++u) {
    float ps[8], pq[8];
#pragma unroll
    for (int v = 0; v < 8; ++v) {
      float s = acc[u][0][v] + acc[u][1][v] + acc[u][2][v] + acc[u][3][v];
      float q = acc[u][0][v] * acc[u][0][v] + acc[u][1][v] * acc[u][1][v] +
                acc[u][2][v] * acc[u][2][v] + acc[u][3][v] * acc[u][3][v];
#pragma unroll
      for (int o = 1; o < 16; o <<= 1) {   // reduce across 16 lanes of this half
        s += __shfl_xor(s, o, 32);
        q += __shfl_xor(q, o, 32);
      }
      ps[v] = s; pq[v] = q;
    }
    if (r == 0) {
#pragma unroll
      for (int v = 0; v < 8; ++v) {
        atomicAdd(&lsum[u * 16 + v + 8 * half], ps[v]);
        atomicAdd(&lsq[u * 16 + v + 8 * half], pq[v]);
      }
    }
  }
  __syncthreads();

  // ---- normalize, scale/shift, ReLU -> LDS (bf16, row-major 32 x 512) ----
#pragma unroll
  for (int t = 0; t < 4; ++t) {
    int h = wave * 64 + t * 16 + r;
    float g = eg[h], bt = ebt[h];
#pragma unroll
    for (int u = 0; u < 2; ++u)
#pragma unroll
      for (int v = 0; v < 8; ++v) {
        int m = u * 16 + v + 8 * half;
        float mu  = lsum[m] * (1.f / H_);
        float var = lsq[m] * (1.f / H_) - mu * mu;
        float x = (acc[u][t][v] - mu) * rsqrtf(var + LN_EPS) * g + bt;
        hn[m * H_ + h] = (__bf16)fmaxf(x, 0.f);
      }
  }
  __syncthreads();

  // ---- GEMM2: out(32x64) = hn(32x512) @ ew2.
  //      wave -> (E-tile = wave&3, K-half = wave>>2), both M-tiles share B.
  v8f c2[2] = {vzero, vzero};
  const int nt2 = wave & 3;
  const int khalf = wave >> 2;
#pragma unroll
  for (int s = 0; s < 8; ++s) {
    int kbase = khalf * 256 + s * 32;
    int k16 = (kbase >> 4) + half;
    const v16bf b2 = *(const v16bf*)(ew2t + (((nt2 * 32) + k16) * 16 + r) * 16);
#pragma unroll
    for (int u = 0; u < 2; ++u) {
      const v16bf a2 = *(const v16bf*)(hn + (u * 16 + r) * H_ + kbase + 16 * half);
      c2[u] = __builtin_amdgcn_wmma_f32_16x16x32_bf16(
          false, a2, false, b2, (short)0, c2[u], false, false);
    }
  }
#pragma unroll
  for (int u = 0; u < 2; ++u)
#pragma unroll
    for (int v = 0; v < 8; ++v) {
      int m = u * 16 + v + 8 * half;
      atomicAdd(&lout[m * E_ + nt2 * 16 + r], c2[u][v]);
    }
  __syncthreads();

  // ---- residual (from the LDS edge tile) + bias, vectorized store ----
  {
    int base = tid * 8;            // 2048 elems / 256 threads
    int m = base >> 6, e = base & 63;
    size_t g = ((size_t)tileBase + m) * E_ + e;
#pragma unroll
    for (int q = 0; q < 2; ++q) {
      float4 ein = *(const float4*)(atile + m * E_ + e + 4 * q);
      float4 o;
      o.x = ein.x + lout[m * E_ + e + 4 * q + 0] + eb2[e + 4 * q + 0];
      o.y = ein.y + lout[m * E_ + e + 4 * q + 1] + eb2[e + 4 * q + 1];
      o.z = ein.z + lout[m * E_ + e + 4 * q + 2] + eb2[e + 4 * q + 2];
      o.w = ein.w + lout[m * E_ + e + 4 * q + 3] + eb2[e + 4 * q + 3];
      *(float4*)(eout + g + 4 * q) = o;
    }
  }
}

extern "C" void kernel_launch(void* const* d_in, const int* in_sizes, int n_in,
                              void* d_out, int out_size, void* d_ws, size_t ws_size,
                              hipStream_t stream) {
  (void)in_sizes; (void)n_in; (void)out_size; (void)ws_size;
  const float* nodes = (const float*)d_in[0];
  const float* edges = (const float*)d_in[1];
  const float* adj   = (const float*)d_in[2];
  const float* nw1   = (const float*)d_in[3];
  const float* nb1   = (const float*)d_in[4];
  const float* ng    = (const float*)d_in[5];
  const float* nbt   = (const float*)d_in[6];
  const float* nw2   = (const float*)d_in[7];
  const float* nb2   = (const float*)d_in[8];
  const float* ew1   = (const float*)d_in[9];
  const float* eb1   = (const float*)d_in[10];
  const float* eg    = (const float*)d_in[11];
  const float* ebt   = (const float*)d_in[12];
  const float* ew2   = (const float*)d_in[13];
  const float* eb2   = (const float*)d_in[14];

  float* out_nodes = (float*)d_out;
  float* out_edges = out_nodes + (size_t)B_ * N_ * D_;

  char* ws = (char*)d_ws;
  __bf16* wet  = (__bf16*)(ws);                                   //  64 KB
  __bf16* ew2t = (__bf16*)(ws + 65536);                           //  64 KB
  float*  xi   = (float*)(ws + 131072);                           // 1.5 MB
  float*  xj   = (float*)(ws + 131072 + 1572864);                 // 1.5 MB
  float*  agg  = (float*)(ws + 131072 + 2 * 1572864);             // 192 KB
  float*  hrel = (float*)(ws + 131072 + 2 * 1572864 + 196608);    // 1.5 MB

  k_pack <<<128,      256, 0, stream>>>(ew1, ew2, wet, ew2t);
  k_xixj <<<B_ * N_,  256, 0, stream>>>(nodes, ew1, xi, xj);
  k_agg  <<<B_ * N_,  256, 0, stream>>>(edges, adj, agg);
  k_node1<<<B_ * N_,  256, 0, stream>>>(agg, nodes, nw1, nb1, ng, nbt, hrel);
  k_node2<<<B_ * N_,  256, 0, stream>>>(nodes, hrel, nw2, nb2, out_nodes);
  k_edge <<<(B_ * N_ * N_) / 32, 256, 0, stream>>>(edges, xi, xj, wet, ew2t,
                                                   eb1, eg, ebt, eb2, out_edges);
}
